// CrossNetwork_39359080300704
// MI455X (gfx1250) — matrix operand order repaired and checked by reading, not compiled, not run on hardware
//
#include <hip/hip_runtime.h>

typedef __attribute__((ext_vector_type(16))) _Float16 v16h;
typedef __attribute__((ext_vector_type(8)))  float    v8f;

#define B_   4096
#define NF   50
#define E_   64
#define T_   2
#define M_   3
#define NFP  64    // NF padded to WMMA tile multiple
#define K2   128   // 2*E
#define LN_EPS 1e-5f
#define NEG_INF_ -1000000000.0f

union Frag { v16h h; unsigned u[8]; };

// ---------------------------------------------------------------------------
// Phase 0: batch-invariant preprocessing (runs once, 1 block).
//  attnT[t][g][f] = attn[t][f][g]  (f16, zero-padded to 64x64)
//  w0T[t][e][k]   = transform_w[0][t][0][k][e]  (f16)
// ---------------------------------------------------------------------------
__global__ __launch_bounds__(256)
void cn_prep(const float* __restrict__ masker,
             const float* __restrict__ tw,
             const float* __restrict__ ln_w,
             const float* __restrict__ ln_b,
             _Float16* __restrict__ attnT,
             _Float16* __restrict__ w0T)
{
    const int tid = threadIdx.x;

    // zero padded attnT
    for (int i = tid; i < T_ * NFP * NFP; i += 256)
        attnT[i] = (_Float16)0.0f;

    // transpose + convert transform_w
    for (int i = tid; i < T_ * E_ * K2; i += 256) {
        int t = i / (E_ * K2);
        int r = i % (E_ * K2);
        int e = r / K2;
        int k = r % K2;
        w0T[i] = (_Float16)tw[(t * K2 + k) * E_ + e];
    }
    __syncthreads();

    // one thread per (t, column j): LayerNorm over rows + masked softmax
    if (tid < T_ * NF) {
        int t = tid / NF;
        int j = tid % NF;
        float adj[NF];
        float mean = 0.0f;
        for (int i = 0; i < NF; ++i) {
            float p = 1.0f;
            for (int m = 0; m < M_; ++m)
                p *= masker[((t * M_ + m) * NF + i) * NF + j];
            float a = p > 0.0f ? p : 0.0f;
            adj[i] = a;
            mean += a;
        }
        mean *= (1.0f / NF);
        float var = 0.0f;
        for (int i = 0; i < NF; ++i) { float d = adj[i] - mean; var += d * d; }
        var *= (1.0f / NF);
        float inv = rsqrtf(var + LN_EPS);

        float lg[NF];
        float mx = -3.0e38f;
        for (int i = 0; i < NF; ++i) {
            float maskv = (adj[i] != 0.0f) ? 1.0f : 0.0f;
            float l = (adj[i] - mean) * inv * ln_w[i] + ln_b[i]
                      + (1.0f - maskv) * NEG_INF_
                      + ((i == j) ? 1.0f : 0.0f);
            lg[i] = l;
            mx = fmaxf(mx, l);
        }
        float s = 0.0f;
        for (int i = 0; i < NF; ++i) { lg[i] = __expf(lg[i] - mx); s += lg[i]; }
        float invs = 1.0f / s;
        for (int i = 0; i < NF; ++i) {
            float maskv = (adj[i] != 0.0f) ? 1.0f : 0.0f;
            attnT[(t * NFP + j) * NFP + i] = (_Float16)(lg[i] * invs * maskv);
        }
    }
}

// ---------------------------------------------------------------------------
// Phase 1: one block (8 waves) per batch item.
// ---------------------------------------------------------------------------
__global__ __launch_bounds__(256)
void cn_main(const float* __restrict__ feat,
             const float* __restrict__ tb,        // (T,1,NF,E)
             const float* __restrict__ gating_w,  // (NF*E,)
             const float* __restrict__ gating_b,  // (1,)
             const _Float16* __restrict__ attnT,  // [T][64][64]
             const _Float16* __restrict__ w0T,    // [T][E][K2]
             float* __restrict__ out)
{
    __shared__ _Float16 s_featT[E_][NFP];   // feat^T  [e][f]  (B operand, nei)
    __shared__ _Float16 s_cat[NFP][K2];     // cat     [f][k]  (A operand, h)
    __shared__ float    s_h[T_][NFP][E_];   // h (f32, pre-gating)
    __shared__ float    s_red[2][256];
    __shared__ float    s_gate[2];

    const int b    = blockIdx.x;
    const int tid  = threadIdx.x;
    const int lane = tid & 31;
    const int wid  = tid >> 5;
    const int lm   = lane & 15;
    const int halfl = lane >> 4;

    // ---- stage feat in both layouts (f16), zero padding rows ----
    for (int i = tid; i < NF * E_; i += 256) {
        int f = i >> 6, e = i & 63;
        _Float16 v = (_Float16)feat[b * (NF * E_) + i];
        s_cat[f][e]   = v;
        s_featT[e][f] = v;
    }
    for (int i = tid; i < E_ * (NFP - NF); i += 256) {
        int e = i / (NFP - NF);
        int f = NF + i % (NFP - NF);
        s_featT[e][f] = (_Float16)0.0f;
        s_cat[f][e]   = (_Float16)0.0f;    // feat half of padded cat rows
    }
    __syncthreads();

    // ---- nei[t] = attn[t]^T (64x64) x feat (64x64), f16 WMMA, f32 acc ----
    // 32 output tiles: (t, mT, nT); each wave owns 4, with nT = wid&3 fixed
    // (B fragments are loop-invariant per wave -> loaded once from LDS).
    {
        const int nT = wid & 3;
        const int n  = nT * 16 + lm;
        Frag Bm0, Bm1;
#pragma unroll
        for (int v = 0; v < 8; ++v) {
            int kk = ((v >> 2) << 4) + halfl * 8 + ((v & 3) << 1);
            Bm0.u[v] = *(const unsigned*)&s_featT[n][kk];
            Bm1.u[v] = *(const unsigned*)&s_featT[n][32 + kk];
        }
#pragma unroll
        for (int it = 0; it < 4; ++it) {
            const int tile = it * 8 + wid;      // uniform per wave, 4 iters
            const int t  = tile >> 4;
            const int mT = (tile >> 2) & 3;
            const int m  = mT * 16 + lm;        // g row
            v8f c = {};
            Frag A0, A1;
            const _Float16* arow = &attnT[(t * NFP + m) * NFP];
#pragma unroll
            for (int v = 0; v < 8; ++v) {
                int kk = ((v >> 2) << 4) + halfl * 8 + ((v & 3) << 1);
                A0.u[v] = *(const unsigned*)&arow[kk];
                A1.u[v] = *(const unsigned*)&arow[32 + kk];
            }
            c = __builtin_amdgcn_wmma_f32_16x16x32_f16(
                    false, A0.h, false, Bm0.h, (short)0, c, false, false);
            c = __builtin_amdgcn_wmma_f32_16x16x32_f16(
                    false, A1.h, false, Bm1.h, (short)0, c, false, false);
            // scatter D into cat's nei half: cat[g][64+e] (f16)
#pragma unroll
            for (int r = 0; r < 8; ++r) {
                int g = mT * 16 + r + halfl * 8;
                s_cat[g][E_ + n] = (_Float16)c[r];
            }
        }
    }
    __syncthreads();

    // ---- h[t] = cat (64x128) x w0[t] (128x64) + bias ----
    {
        const int nT = wid & 3;
        const int n  = nT * 16 + lm;            // e col
#pragma unroll
        for (int it = 0; it < 4; ++it) {
            const int tile = it * 8 + wid;
            const int t  = tile >> 4;
            const int mT = (tile >> 2) & 3;
            const int m  = mT * 16 + lm;        // f row
            v8f c = {};
            const _Float16* brow = &w0T[(t * E_ + n) * K2];
#pragma unroll
            for (int kc = 0; kc < 4; ++kc) {
                const int kBase = kc * 32;
                Frag A, Bm;
#pragma unroll
                for (int v = 0; v < 8; ++v) {
                    int kk = ((v >> 2) << 4) + halfl * 8 + ((v & 3) << 1);
                    A.u[v] = *(const unsigned*)&s_cat[m][kBase + kk];
                }
#pragma unroll
                for (int v = 0; v < 8; ++v) {
                    int kk = ((v >> 2) << 4) + halfl * 8 + ((v & 3) << 1);
                    Bm.u[v] = *(const unsigned*)&brow[kBase + kk];
                }
                c = __builtin_amdgcn_wmma_f32_16x16x32_f16(
                        false, A.h, false, Bm.h, (short)0, c, false, false);
            }
#pragma unroll
            for (int r = 0; r < 8; ++r) {
                int f = mT * 16 + r + halfl * 8;
                if (f < NF)
                    s_h[t][f][n] = c[r] + tb[(t * NF + f) * E_ + n];
            }
        }
    }
    __syncthreads();

    // ---- gating: logit[t] = <h[t], gw> + gb; softmax over T=2 ----
    float p0 = 0.0f, p1 = 0.0f;
    for (int i = tid; i < NF * E_; i += 256) {
        int f = i >> 6, e = i & 63;
        float gw = gating_w[i];
        p0 += s_h[0][f][e] * gw;
        p1 += s_h[1][f][e] * gw;
    }
    s_red[0][tid] = p0;
    s_red[1][tid] = p1;
    __syncthreads();
    for (int s = 128; s > 0; s >>= 1) {
        if (tid < s) {
            s_red[0][tid] += s_red[0][tid + s];
            s_red[1][tid] += s_red[1][tid + s];
        }
        __syncthreads();
    }
    if (tid == 0) {
        float gb = gating_b[0];
        float l0 = s_red[0][0] + gb;
        float l1 = s_red[1][0] + gb;
        float mx = fmaxf(l0, l1);
        float e0 = __expf(l0 - mx), e1 = __expf(l1 - mx);
        float inv = 1.0f / (e0 + e1);
        s_gate[0] = e0 * inv;
        s_gate[1] = e1 * inv;
    }
    __syncthreads();

    float g0 = s_gate[0], g1 = s_gate[1];
    for (int i = tid; i < NF * E_; i += 256) {
        int f = i >> 6, e = i & 63;
        out[b * (NF * E_) + i] = g0 * s_h[0][f][e] + g1 * s_h[1][f][e];
    }
}

// ---------------------------------------------------------------------------
extern "C" void kernel_launch(void* const* d_in, const int* in_sizes, int n_in,
                              void* d_out, int out_size, void* d_ws, size_t ws_size,
                              hipStream_t stream)
{
    const float* feat      = (const float*)d_in[0];
    const float* masker    = (const float*)d_in[1];
    const float* tw        = (const float*)d_in[2];
    const float* tb        = (const float*)d_in[3];
    const float* ln_w      = (const float*)d_in[4];
    const float* ln_b      = (const float*)d_in[5];
    const float* gating_w  = (const float*)d_in[6];
    const float* gating_b  = (const float*)d_in[7];

    _Float16* attnT = (_Float16*)d_ws;                 // T*64*64 f16 = 16 KB
    _Float16* w0T   = attnT + T_ * NFP * NFP;          // T*64*128 f16 = 32 KB

    cn_prep<<<1, 256, 0, stream>>>(masker, tw, ln_w, ln_b, attnT, w0T);
    cn_main<<<B_, 256, 0, stream>>>(feat, tb, gating_w, gating_b,
                                    attnT, w0T, (float*)d_out);
}
